// SASA_29472065585620
// MI455X (gfx1250) — compile-verified, matrix operand order checked by reading, not run on hardware
//
#include <hip/hip_runtime.h>
#include <hip/hip_bf16.h>
#include <stdint.h>

#define CDIM 41
#define BDIM 8192
#define TDIM 32
#define NSEG 5
#define MDIM 200                 // (C-1)*NSEG
#define TC   (TDIM*CDIM)         // 1312
#define CB   (CDIM*BDIM)         // 335872
#define SB   (NSEG*BDIM)         // 40960

typedef __attribute__((ext_vector_type(16))) _Float16 v16h;
typedef __attribute__((ext_vector_type(8)))  float    v8f;

__device__ __forceinline__ float sigf(float x){ return 1.0f/(1.0f + __expf(-x)); }
__device__ __forceinline__ float tanh_(float x){ float e = __expf(2.0f*x); return 1.0f - 2.0f/(e + 1.0f); }
__device__ __forceinline__ float wred_sum(float v){
  #pragma unroll
  for (int o = 16; o > 0; o >>= 1) v += __shfl_xor(v, o, 32);
  return v;
}
__device__ __forceinline__ float wred_max(float v){
  #pragma unroll
  for (int o = 16; o > 0; o >>= 1) v = fmaxf(v, __shfl_xor(v, o, 32));
  return v;
}

// ---------------------------------------------------------------------------
// Kernel 1: 5-window per-channel LSTM. One thread = one (c,b).
// x column (strided) is staged via CDNA5 async global->LDS DMA, overlapped
// with weight loads; hidden/cell states for the 5 windows live in registers.
// Writes h_all[c][s][b] as float2 to workspace.
// ---------------------------------------------------------------------------
__global__ __launch_bounds__(256) void k_lstm(
    const float* __restrict__ x,   const float* __restrict__ w_ih,
    const float* __restrict__ w_hh,const float* __restrict__ b_ih,
    const float* __restrict__ b_hh,float* __restrict__ hall)
{
  __shared__ float xs[20][256];
  const int g = blockIdx.x * 256 + threadIdx.x;
  const int c = g >> 13;          // g / 8192
  const int b = g & (BDIM - 1);

  // async-stage x[b, t=12..31, c] into LDS (CDNA5 ASYNCcnt path)
  {
    const float* gp0 = x + (size_t)b * TC + 12 * CDIM + c;
    uint32_t l0 = (uint32_t)(uintptr_t)(&xs[0][threadIdx.x]);
    #pragma unroll
    for (int tt = 0; tt < 20; ++tt) {
      unsigned long long ga = (unsigned long long)(uintptr_t)(gp0 + tt * CDIM);
      uint32_t la = l0 + (uint32_t)(tt * 256 * 4);
      asm volatile("global_load_async_to_lds_b32 %0, %1, off"
                   :: "v"(la), "v"(ga) : "memory");
    }
  }

  // per-channel weights (overlap with async loads in flight)
  float wih[8], whh[16], bs[8];
  #pragma unroll
  for (int j = 0; j < 8; ++j) { wih[j] = w_ih[c*8+j]; bs[j] = b_ih[c*8+j] + b_hh[c*8+j]; }
  #pragma unroll
  for (int j = 0; j < 16; ++j) whh[j] = w_hh[c*16+j];

  float hs[NSEG][2], cs[NSEG][2];
  #pragma unroll
  for (int s = 0; s < NSEG; ++s) { hs[s][0]=0.f; hs[s][1]=0.f; cs[s][0]=0.f; cs[s][1]=0.f; }

  asm volatile("s_wait_asynccnt 0" ::: "memory");

  for (int t = 12; t < 32; ++t) {
    float xt = xs[t - 12][threadIdx.x];
    float xg[8];
    #pragma unroll
    for (int j = 0; j < 8; ++j) xg[j] = fmaf(xt, wih[j], bs[j]);
    #pragma unroll
    for (int s = 0; s < NSEG; ++s) {
      if (t >= 28 - 4*s) {              // window s active from start = 28-4s
        float h0 = hs[s][0], h1 = hs[s][1];
        float gg[8];
        #pragma unroll
        for (int j = 0; j < 8; ++j)
          gg[j] = fmaf(h0, whh[2*j], fmaf(h1, whh[2*j+1], xg[j]));
        float i0=sigf(gg[0]), i1=sigf(gg[1]);
        float f0=sigf(gg[2]), f1=sigf(gg[3]);
        float g0=tanh_(gg[4]), g1=tanh_(gg[5]);
        float o0=sigf(gg[6]), o1=sigf(gg[7]);
        float c0 = fmaf(f0, cs[s][0], i0*g0);
        float c1 = fmaf(f1, cs[s][1], i1*g1);
        cs[s][0]=c0; cs[s][1]=c1;
        hs[s][0]=o0*tanh_(c0); hs[s][1]=o1*tanh_(c1);
      }
    }
  }
  float2* hp = (float2*)hall;
  #pragma unroll
  for (int s = 0; s < NSEG; ++s)
    hp[(size_t)(c*NSEG + s)*BDIM + b] = make_float2(hs[s][0], hs[s][1]);
}

// ---------------------------------------------------------------------------
// Kernel 2: intra-channel attention. One thread = one (c,b'). The 160 gathered
// hidden vectors per wave are projected to Q|K|V by v_wmma_f32_16x16x32_f16
// (K padded 2->32, B = packed [wq|wk|wv]^T). mean_t(Q.K_t) = Q . mean_t(K_t).
// Exact sparsemax over 5. Writes att_intra and Z.
// ---------------------------------------------------------------------------
__global__ __launch_bounds__(256) void k_intra(
    const float* __restrict__ hall,
    const float* __restrict__ wq, const float* __restrict__ bq,
    const float* __restrict__ wk, const float* __restrict__ bk,
    const float* __restrict__ wv, const float* __restrict__ bv,
    float* __restrict__ att_intra, float* __restrict__ zbuf)
{
  __shared__ float hbuf[8][160][2];
  __shared__ float qkv[8][160][6];
  const int g    = blockIdx.x * 256 + threadIdx.x;
  const int c    = g >> 13;
  const int bp   = g & (BDIM - 1);
  const int w    = threadIdx.x >> 5;
  const int lane = threadIdx.x & 31;
  const float2* hp = (const float2*)hall;

  // scrambled gather: raw torch reshape [S,B,h] -> [B,S,h]
  #pragma unroll
  for (int sp = 0; sp < NSEG; ++sp) {
    int fl   = bp * NSEG + sp;
    int ssrc = fl >> 13;
    int bsrc = fl & (BDIM - 1);
    float2 hv = hp[(size_t)(c*NSEG + ssrc)*BDIM + bsrc];
    hbuf[w][lane*NSEG + sp][0] = hv.x;
    hbuf[w][lane*NSEG + sp][1] = hv.y;
  }

  // B operand: Wcat (2 x 16), cols 0-1 = wq^T, 2-3 = wk^T, 4-5 = wv^T
  v16h bm = {};
  {
    float w0 = 0.f, w1 = 0.f;
    int n = lane;
    if (n < 2)      { w0 = wq[n*2+0];     w1 = wq[n*2+1];     }
    else if (n < 4) { w0 = wk[(n-2)*2+0]; w1 = wk[(n-2)*2+1]; }
    else if (n < 6) { w0 = wv[(n-4)*2+0]; w1 = wv[(n-4)*2+1]; }
    bm[0] = (_Float16)w0; bm[1] = (_Float16)w1;   // lanes>=6 / >=16 stay zero
  }
  asm volatile("" ::: "memory");   // keep LDS writes before intra-wave reads

  // 10 WMMAs per wave: project 160 h-vectors to 6 outputs each
  #pragma unroll
  for (int q = 0; q < 10; ++q) {
    v16h a = {};
    if (lane < 16) {
      a[0] = (_Float16)hbuf[w][q*16 + lane][0];   // K=0
      a[1] = (_Float16)hbuf[w][q*16 + lane][1];   // K=1
    }
    v8f acc = {};
    acc = __builtin_amdgcn_wmma_f32_16x16x32_f16(false, a, false, bm,
                                                 (short)0, acc, false, false);
    int nn = lane & 15, up = lane >> 4;
    if (nn < 6) {
      #pragma unroll
      for (int r = 0; r < 8; ++r)
        qkv[w][q*16 + r + up*8][nn] = acc[r];     // D: VGPR r = rows r / r+8
    }
  }
  asm volatile("" ::: "memory");

  const float bq0=bq[0], bq1=bq[1], bk0=bk[0], bk1=bk[1], bv0=bv[0], bv1=bv[1];
  float Qx[5],Qy[5],Vx[5],Vy[5], Kbx=0.f, Kby=0.f;
  #pragma unroll
  for (int s = 0; s < 5; ++s) {
    float* row = qkv[w][lane*5 + s];
    Qx[s] = row[0] + bq0;  Qy[s] = row[1] + bq1;
    Kbx  += row[2] + bk0;  Kby  += row[3] + bk1;
    Vx[s] = row[4] + bv0;  Vy[s] = row[5] + bv1;
  }
  Kbx *= 0.2f; Kby *= 0.2f;                       // mean over t segments

  float z[5];
  #pragma unroll
  for (int s = 0; s < 5; ++s)
    z[s] = (Qx[s]*Kbx + Qy[s]*Kby) * 0.7071067811865475f;   // / sqrt(h)

  // exact sparsemax over 5
  float zmax = fmaxf(fmaxf(fmaxf(z[0],z[1]), fmaxf(z[2],z[3])), z[4]);
  #pragma unroll
  for (int s = 0; s < 5; ++s) z[s] -= zmax;
  float zs[5] = {z[0], z[1], z[2], z[3], z[4]};
  #define CSWAP(i,j) { float mx = fmaxf(zs[i],zs[j]); zs[j] = fminf(zs[i],zs[j]); zs[i] = mx; }
  CSWAP(0,1) CSWAP(3,4) CSWAP(2,4) CSWAP(2,3) CSWAP(1,4)
  CSWAP(0,3) CSWAP(0,2) CSWAP(1,3) CSWAP(1,2)
  #undef CSWAP
  float cum = 0.f, kk = 1.f, ss = zs[0];
  #pragma unroll
  for (int r = 1; r <= 5; ++r) {
    cum += zs[r-1];
    if (1.f + (float)r * zs[r-1] > cum) { kk = (float)r; ss = cum; }
  }
  float tau = (ss - 1.f) / kk;
  float Zx = 0.f, Zy = 0.f;
  #pragma unroll
  for (int s = 0; s < 5; ++s) {
    float p = fmaxf(z[s] - tau, 0.f);
    att_intra[(size_t)c*BDIM*NSEG + (size_t)bp*NSEG + s] = p;
    Zx = fmaf(p, Vx[s], Zx); Zy = fmaf(p, Vy[s], Zy);
  }
  zbuf[(size_t)g*2 + 0] = Zx * 0.2f;
  zbuf[(size_t)g*2 + 1] = Zy * 0.2f;
}

// ---------------------------------------------------------------------------
// Kernel 3: inter-channel attention. One wave32 per (c,b'); 200 logits spread
// 7-per-lane; sparsemax-200 via wave-cooperative bisection + exact support
// refinement. Writes att_inter (dominant traffic) and H.
// ---------------------------------------------------------------------------
__global__ __launch_bounds__(256) void k_inter(
    const float* __restrict__ hall, const float* __restrict__ zbuf,
    float* __restrict__ att_inter, float* __restrict__ Hout)
{
  const int wg   = blockIdx.x * 8 + (threadIdx.x >> 5);
  const int lane = threadIdx.x & 31;
  const int c    = wg >> 13;
  const int bp   = wg & (BDIM - 1);
  const float2* hp = (const float2*)hall;

  float zx = zbuf[(size_t)wg*2 + 0], zy = zbuf[(size_t)wg*2 + 1];
  float zn  = fmaxf(sqrtf(zx*zx + zy*zy), 1e-12f);
  float nzx = zx / zn, nzy = zy / zn;

  float ox[7], oy[7], d[7];
  #pragma unroll
  for (int k = 0; k < 7; ++k) {
    int m = lane + 32*k;
    ox[k] = 0.f; oy[k] = 0.f; d[k] = -1e30f;
    if (m < MDIM) {
      unsigned fl   = (unsigned)bp * MDIM + (unsigned)m;   // raw reshape scramble
      unsigned jidx = fl / SB;
      unsigned rem  = fl - jidx * SB;
      unsigned ssrc = rem >> 13;
      unsigned bsrc = rem & (BDIM - 1);
      int j = (int)jidx + ((int)jidx >= c ? 1 : 0);        // skip own channel
      float2 o = hp[(size_t)(j*NSEG + (int)ssrc)*BDIM + bsrc];
      ox[k] = o.x; oy[k] = o.y;
      float on = fmaxf(sqrtf(o.x*o.x + o.y*o.y), 1e-12f);
      d[k] = (nzx*o.x + nzy*o.y) / on;
    }
  }

  float lm = -1e30f;
  #pragma unroll
  for (int k = 0; k < 7; ++k) lm = fmaxf(lm, d[k]);
  float zmax = wred_max(lm);
  #pragma unroll
  for (int k = 0; k < 7; ++k) if (d[k] > -1e29f) d[k] -= zmax;

  // bisection for tau: sum(max(z - tau, 0)) = 1, tau in [-1, 0]
  float lo = -1.f, hi = 0.f;
  for (int it = 0; it < 26; ++it) {
    float mid = 0.5f*(lo + hi), ls = 0.f;
    #pragma unroll
    for (int k = 0; k < 7; ++k) ls += fmaxf(d[k] - mid, 0.f);
    float tot = wred_sum(ls);
    if (tot >= 1.f) lo = mid; else hi = mid;
  }
  float tmid = 0.5f*(lo + hi), lk = 0.f, lsum = 0.f;
  #pragma unroll
  for (int k = 0; k < 7; ++k) if (d[k] > tmid) { lk += 1.f; lsum += d[k]; }
  float kk   = wred_sum(lk);
  float ssum = wred_sum(lsum);
  float tau  = (ssum - 1.f) / kk;                 // exact on identified support

  float ux = 0.f, uy = 0.f;
  float* ai = att_inter + (size_t)c*BDIM*MDIM + (size_t)bp*MDIM;
  #pragma unroll
  for (int k = 0; k < 7; ++k) {
    int m = lane + 32*k;
    if (m < MDIM) {
      float pv = fmaxf(d[k] - tau, 0.f);
      ai[m] = pv;
      ux = fmaf(pv, ox[k], ux); uy = fmaf(pv, oy[k], uy);
    }
  }
  ux = wred_sum(ux) * (1.0f/MDIM);
  uy = wred_sum(uy) * (1.0f/MDIM);
  if (lane == 0)
    *(float4*)(Hout + (size_t)bp*(CDIM*4) + c*4) = make_float4(zx, zy, ux, uy);
}

extern "C" void kernel_launch(void* const* d_in, const int* in_sizes, int n_in,
                              void* d_out, int out_size, void* d_ws, size_t ws_size,
                              hipStream_t stream) {
  (void)in_sizes; (void)n_in; (void)out_size; (void)ws_size;
  const float* x    = (const float*)d_in[0];
  const float* w_ih = (const float*)d_in[1];
  const float* w_hh = (const float*)d_in[2];
  const float* b_ih = (const float*)d_in[3];
  const float* b_hh = (const float*)d_in[4];
  const float* wq   = (const float*)d_in[5];
  const float* bq   = (const float*)d_in[6];
  const float* wk   = (const float*)d_in[7];
  const float* bk   = (const float*)d_in[8];
  const float* wv   = (const float*)d_in[9];
  const float* bv   = (const float*)d_in[10];

  float* hall = (float*)d_ws;                                  // [C,S,B,2] 13.4 MB
  float* zbuf = hall + (size_t)CDIM*NSEG*BDIM*2;               // [C,B,2]    2.7 MB

  float* out       = (float*)d_out;
  float* att_intra = out;                                      // C*B*S
  float* att_inter = out + (size_t)CDIM*BDIM*NSEG;             // C*B*M
  float* Hout      = att_inter + (size_t)CDIM*BDIM*MDIM;       // B*C*4

  dim3 blk(256);
  k_lstm <<<CB/256, blk, 0, stream>>>(x, w_ih, w_hh, b_ih, b_hh, hall);
  k_intra<<<CB/256, blk, 0, stream>>>(hall, wq, bq, wk, bk, wv, bv, att_intra, zbuf);
  k_inter<<<CB/8,   blk, 0, stream>>>(hall, zbuf, att_inter, Hout);
}
